// MambaBlockV2_37263136260203
// MI455X (gfx1250) — compile-verified, matrix operand order checked by reading, not run on hardware
//
#include <hip/hip_runtime.h>
#include <hip/hip_bf16.h>

// ---------------------------------------------------------------------------
// Bidirectional Mamba block for gfx1250 (MI455X).
//   - bf16 WMMA (v_wmma_f32_16x16x32_bf16) for in_proj / x_proj / out_proj
//   - GEMM: 8-wave blocks, W slab staged to LDS via async global->LDS copy
//     (double buffered, ASYNCcnt), A fragments software-pipelined in VGPRs
//   - fused conv1d(d=4)+SiLU
//   - selective scan: 8192 parallel channels, LDS-staged per-step B/C/dt_low,
//     dt projection (K=16) + softplus + D-skip + SiLU(z) gating fused in
// ---------------------------------------------------------------------------

#define BATCH_   8
#define SEQ_     2048
#define DMODEL_  256
#define DINNER_  1024
#define DSTATE_  16
#define DTRANK_  16
#define MROWS_   (BATCH_ * SEQ_)          // 16384

typedef __attribute__((ext_vector_type(16))) __bf16 v16bf;
typedef __attribute__((ext_vector_type(8)))  __bf16 v8bf;
typedef __attribute__((ext_vector_type(8)))  float  v8f;
typedef __attribute__((ext_vector_type(4)))  int    v4i;
typedef __attribute__((address_space(1))) v4i as1_v4i;   // global
typedef __attribute__((address_space(3))) v4i as3_v4i;   // LDS

#ifndef __has_builtin
#define __has_builtin(x) 0
#endif
#if __has_builtin(__builtin_amdgcn_global_load_async_to_lds_b128) && \
    __has_builtin(__builtin_amdgcn_s_wait_asynccnt)
#define USE_ASYNC_LDS 1
#else
#define USE_ASYNC_LDS 0
#endif

static __device__ __forceinline__ __bf16 f2bf(float f) {
    unsigned u = __builtin_bit_cast(unsigned, f);
    unsigned r = u + 0x7FFFu + ((u >> 16) & 1u);          // round-to-nearest-even
    return __builtin_bit_cast(__bf16, (unsigned short)(r >> 16));
}

static __device__ __forceinline__ v16bf cat16(v8bf lo, v8bf hi) {
    return __builtin_shufflevector(lo, hi, 0,1,2,3,4,5,6,7,8,9,10,11,12,13,14,15);
}

// 16B global -> LDS copy; async (ASYNCcnt) when the builtin is available.
// AS pointers are formed via integer casts: generic LDS addr low 32 bits are
// the LDS offset (ISA 10.2); global generic == AS1 numerically.
static __device__ __forceinline__ void cp16_to_lds(const __bf16* g, __bf16* l) {
#if USE_ASYNC_LDS
    __builtin_amdgcn_global_load_async_to_lds_b128(
        (as1_v4i*)(unsigned long long)g,
        (as3_v4i*)(unsigned int)(unsigned long long)l,
        0, 0);
#else
    *(v8bf*)l = *(const v8bf*)g;
#endif
}

// --------------------------- f32 -> bf16 convert ---------------------------
__global__ void cvt_bf16_kernel(const float* __restrict__ in,
                                __bf16* __restrict__ out, int n) {
    int i = blockIdx.x * blockDim.x + threadIdx.x;
    if (i < n) out[i] = f2bf(in[i]);
}

// ------------------------------- RMSNorm -----------------------------------
// one wave (32 lanes) per row of 256; optional time-flip on the read side
__global__ void rmsnorm_kernel(const float* __restrict__ x,
                               const float* __restrict__ w,
                               __bf16* __restrict__ xn, int rev) {
    int wave = threadIdx.x >> 5, lane = threadIdx.x & 31;
    long row = (long)blockIdx.x * 8 + wave;               // [0, MROWS_)
    int b = (int)(row / SEQ_), l = (int)(row % SEQ_);
    int lo = rev ? (SEQ_ - 1 - l) : l;
    const float* xp = x + ((long)b * SEQ_ + lo) * DMODEL_ + lane * 8;
    float v[8], s = 0.f;
#pragma unroll
    for (int i = 0; i < 8; ++i) { v[i] = xp[i]; s += v[i] * v[i]; }
#pragma unroll
    for (int off = 16; off >= 1; off >>= 1) s += __shfl_xor(s, off, 32);
    float scale = rsqrtf(s * (1.f / DMODEL_) + 1e-5f);
    __bf16* op = xn + row * DMODEL_ + lane * 8;
#pragma unroll
    for (int i = 0; i < 8; ++i) op[i] = f2bf(v[i] * scale * w[lane * 8 + i]);
}

// ------------------------- generic bf16 WMMA GEMM --------------------------
// C[M,N] (f32) = A[M,K] (bf16, K-contig) * W[N,K]^T (bf16, K-contig)
// Block = 8 waves covering a 128(M) x 16(N) tile; the 16x32 W slab is staged
// into LDS (async global->LDS, double buffered) and shared by all 8 waves.
// A fragments are pipelined in registers one K-step ahead.
__global__ __launch_bounds__(256)
void gemm_bf16_wmma_kernel(const __bf16* __restrict__ A,
                           const __bf16* __restrict__ W,
                           float* __restrict__ C,
                           int M, int N, int K) {
    __shared__ __bf16 sW[2][16 * 32];                 // 2 x 1 KB double buffer

    const int tid  = threadIdx.x;
    const int wv   = tid >> 5;
    const int lane = tid & 31;
    const int r    = lane & 15, half = lane >> 4;
    const int tn   = blockIdx.x;                      // N tile (16 cols)
    const int tm   = blockIdx.y * 8 + wv;             // M tile (16 rows) per wave

    // W slab staging: 16 rows x 32 bf16 = 64 chunks of 16B; threads 0..63
    const int srow = tid >> 2, schunk = tid & 3;
    const __bf16* wsrc = W + (size_t)(tn * 16 + srow) * K + schunk * 8;
    __bf16* sdst0 = &sW[0][srow * 32 + schunk * 8];
    __bf16* sdst1 = &sW[1][srow * 32 + schunk * 8];

    const __bf16* Ap = A + (size_t)(tm * 16 + r) * K + half * 8;

    if (tid < 64) cp16_to_lds(wsrc, sdst0);           // prologue: slab 0
    v8bf a0 = *(const v8bf*)(Ap);
    v8bf a1 = *(const v8bf*)(Ap + 16);
    v8f  c  = {};
    int  p  = 0;

    for (int k0 = 0; k0 < K; k0 += 32) {
        const int kn = k0 + 32;
        if (kn < K && tid < 64)                       // stage next slab
            cp16_to_lds(wsrc + kn, p ? sdst0 : sdst1);
#if USE_ASYNC_LDS
        if (kn < K) __builtin_amdgcn_s_wait_asynccnt(1);  // current slab landed
        else        __builtin_amdgcn_s_wait_asynccnt(0);
#endif
        __syncthreads();                                  // sW[p] visible to all

        v8bf na0 = a0, na1 = a1;                          // pipeline next A frag
        if (kn < K) {
            na0 = *(const v8bf*)(Ap + kn);
            na1 = *(const v8bf*)(Ap + kn + 16);
        }
        v8bf b0 = *(const v8bf*)&sW[p][r * 32 + half * 8];
        v8bf b1 = *(const v8bf*)&sW[p][r * 32 + 16 + half * 8];
        c = __builtin_amdgcn_wmma_f32_16x16x32_bf16(false, cat16(a0, a1),
                                                    false, cat16(b0, b1),
                                                    (short)0, c, false, false);
        a0 = na0; a1 = na1;
        __syncthreads();                                  // done reading sW[p]
        p ^= 1;
    }

    // f32 C/D layout: VGPR i -> M = i + half*8, N = lane&15
    float* Cp = C + (size_t)(tm * 16 + half * 8) * N + tn * 16 + r;
#pragma unroll
    for (int i = 0; i < 8; ++i) Cp[(size_t)i * N] = c[i];
}

// ----------------------- depthwise causal conv + SiLU ----------------------
__global__ void conv_silu_kernel(const float* __restrict__ xz,
                                 const float* __restrict__ cw,
                                 const float* __restrict__ cb,
                                 float* __restrict__ xc32,
                                 __bf16* __restrict__ xc16) {
    long idx = (long)blockIdx.x * blockDim.x + threadIdx.x;
    if (idx >= (long)MROWS_ * DINNER_) return;
    int d  = (int)(idx & (DINNER_ - 1));
    long m = idx >> 10;
    int l  = (int)(m % SEQ_);
    long mb = m - l;                              // b*SEQ_
    float acc = cb[d];
#pragma unroll
    for (int j = 0; j < 4; ++j) {
        int ll = l - 3 + j;
        if (ll >= 0) acc += cw[d * 4 + j] * xz[(mb + ll) * (2 * DINNER_) + d];
    }
    float s = acc / (1.f + __expf(-acc));         // SiLU
    xc32[idx] = s;
    xc16[idx] = f2bf(s);
}

// ------------------------------ selective scan -----------------------------
// block = 256 threads = one (batch, 256-channel group); LDS stages the 48
// shared values (dt_low[16], B[16], C[16]) per timestep.  Fuses dt projection
// + softplus, D skip-connection and SiLU(z) gating.  Emits y in bf16.
__global__ void scan_kernel(const float* __restrict__ dbl,
                            const float* __restrict__ xc32,
                            const float* __restrict__ xz,
                            const float* __restrict__ dt_w,
                            const float* __restrict__ dt_b,
                            const float* __restrict__ A_log,
                            const float* __restrict__ Dp,
                            __bf16* __restrict__ y16) {
    int tid = threadIdx.x;
    int b = blockIdx.x >> 2;
    int d = ((blockIdx.x & 3) << 8) + tid;

    float Arow[DSTATE_], dtw[DTRANK_], h[DSTATE_];
#pragma unroll
    for (int n = 0; n < DSTATE_; ++n) {
        Arow[n] = -__expf(A_log[d * DSTATE_ + n]);
        h[n] = 0.f;
    }
#pragma unroll
    for (int r = 0; r < DTRANK_; ++r) dtw[r] = dt_w[d * DTRANK_ + r];
    float dtb = dt_b[d], Dd = Dp[d];

    __shared__ float sD[48];
    long mbase = (long)b * SEQ_;
    for (int l = 0; l < SEQ_; ++l) {
        long m = mbase + l;
        if (tid < 48) sD[tid] = dbl[m * 48 + tid];
        __syncthreads();
        float dtp = dtb;
#pragma unroll
        for (int r = 0; r < DTRANK_; ++r) dtp += sD[r] * dtw[r];
        float dt = dtp > 20.f ? dtp : __logf(1.f + __expf(dtp));   // softplus
        float u = xc32[m * DINNER_ + d];
        float z = xz[m * (2 * DINNER_) + DINNER_ + d];
        float du = dt * u;
        float y = 0.f;
#pragma unroll
        for (int n = 0; n < DSTATE_; ++n) {
            float dA = __expf(dt * Arow[n]);
            h[n] = h[n] * dA + du * sD[16 + n];
            y += h[n] * sD[32 + n];
        }
        y += u * Dd;
        y *= z / (1.f + __expf(-z));              // * SiLU(z)
        y16[m * DINNER_ + d] = f2bf(y);
        __syncthreads();
    }
}

// ---------------------- residual add + (un)flip + concat -------------------
__global__ void epilogue_kernel(const float* __restrict__ yo,
                                const float* __restrict__ x,
                                float* __restrict__ out, int rev, int off) {
    long idx = (long)blockIdx.x * blockDim.x + threadIdx.x;
    if (idx >= (long)MROWS_ * DMODEL_) return;
    int c  = (int)(idx & (DMODEL_ - 1));
    long m = idx >> 8;
    int b = (int)(m / SEQ_), l = (int)(m % SEQ_);
    int lo = rev ? (SEQ_ - 1 - l) : l;
    long mo = (long)b * SEQ_ + lo;
    out[mo * (2 * DMODEL_) + off + c] = yo[m * DMODEL_ + c] + x[mo * DMODEL_ + c];
}

// ---------------------------------------------------------------------------
extern "C" void kernel_launch(void* const* d_in, const int* in_sizes, int n_in,
                              void* d_out, int out_size, void* d_ws, size_t ws_size,
                              hipStream_t stream) {
    (void)in_sizes; (void)n_in; (void)out_size; (void)ws_size;
    const float* x = (const float*)d_in[0];
    float* out = (float*)d_out;

    // workspace carve-up (bytes, 256-aligned), reused across the 2 directions
    char* ws = (char*)d_ws;
    size_t o = 0;
    float*  xz    = (float*)(ws + o);  o += (size_t)MROWS_ * 2 * DINNER_ * 4;  // 134 MB
    float*  xc32  = (float*)(ws + o);  o += (size_t)MROWS_ * DINNER_ * 4;      //  67 MB
    __bf16* xc16  = (__bf16*)(ws + o); o += (size_t)MROWS_ * DINNER_ * 2;      //  33 MB
    float*  dbl   = (float*)(ws + o);  o += (size_t)MROWS_ * 48 * 4;           //   3 MB
    __bf16* y16   = (__bf16*)(ws + o); o += (size_t)MROWS_ * DINNER_ * 2;      //  33 MB
    float*  yo32  = (float*)(ws + o);  o += (size_t)MROWS_ * DMODEL_ * 4;      //  17 MB
    __bf16* xn16  = (__bf16*)(ws + o); o += (size_t)MROWS_ * DMODEL_ * 2;      //   8 MB
    __bf16* win16 = (__bf16*)(ws + o); o += (size_t)2 * DINNER_ * DMODEL_ * 2; //   1 MB
    __bf16* wxp16 = (__bf16*)(ws + o); o += (size_t)48 * DINNER_ * 2;
    __bf16* wout16= (__bf16*)(ws + o); o += (size_t)DMODEL_ * DINNER_ * 2;

    for (int rev = 0; rev < 2; ++rev) {
        int base = 1 + rev * 10;
        const float* norm_w   = (const float*)d_in[base + 0];
        const float* in_proj  = (const float*)d_in[base + 1];
        const float* conv_w   = (const float*)d_in[base + 2];
        const float* conv_b   = (const float*)d_in[base + 3];
        const float* x_proj   = (const float*)d_in[base + 4];
        const float* dt_w     = (const float*)d_in[base + 5];
        const float* dt_b     = (const float*)d_in[base + 6];
        const float* A_log    = (const float*)d_in[base + 7];
        const float* Dvec     = (const float*)d_in[base + 8];
        const float* out_proj = (const float*)d_in[base + 9];

        // weights -> bf16
        int nw;
        nw = 2 * DINNER_ * DMODEL_;
        cvt_bf16_kernel<<<(nw + 255) / 256, 256, 0, stream>>>(in_proj, win16, nw);
        nw = 48 * DINNER_;
        cvt_bf16_kernel<<<(nw + 255) / 256, 256, 0, stream>>>(x_proj, wxp16, nw);
        nw = DMODEL_ * DINNER_;
        cvt_bf16_kernel<<<(nw + 255) / 256, 256, 0, stream>>>(out_proj, wout16, nw);

        // RMSNorm (+flip for bw) -> bf16 activations
        rmsnorm_kernel<<<MROWS_ / 8, 256, 0, stream>>>(x, norm_w, xn16, rev);

        // in_proj: [M,256] x [2048,256]^T -> xz [M,2048]
        gemm_bf16_wmma_kernel<<<dim3(2 * DINNER_ / 16, MROWS_ / 128), 256, 0, stream>>>(
            xn16, win16, xz, MROWS_, 2 * DINNER_, DMODEL_);

        // depthwise causal conv + SiLU
        {
            long n = (long)MROWS_ * DINNER_;
            conv_silu_kernel<<<(unsigned)((n + 255) / 256), 256, 0, stream>>>(
                xz, conv_w, conv_b, xc32, xc16);
        }

        // x_proj: [M,1024] x [48,1024]^T -> dbl [M,48]
        gemm_bf16_wmma_kernel<<<dim3(48 / 16, MROWS_ / 128), 256, 0, stream>>>(
            xc16, wxp16, dbl, MROWS_, 48, DINNER_);

        // selective scan (fused dt proj + softplus + D skip + SiLU gate)
        scan_kernel<<<BATCH_ * (DINNER_ / 256), 256, 0, stream>>>(
            dbl, xc32, xz, dt_w, dt_b, A_log, Dvec, y16);

        // out_proj: [M,1024] x [256,1024]^T -> yo32 [M,256]
        gemm_bf16_wmma_kernel<<<dim3(DMODEL_ / 16, MROWS_ / 128), 256, 0, stream>>>(
            y16, wout16, yo32, MROWS_, DMODEL_, DINNER_);

        // residual + unflip + concat into d_out [B,L,512]
        {
            long n = (long)MROWS_ * DMODEL_;
            epilogue_kernel<<<(unsigned)((n + 255) / 256), 256, 0, stream>>>(
                yo32, x, out, rev, rev * DMODEL_);
        }
    }
}
